// PointNetPlusPartseg_61323543052858
// MI455X (gfx1250) — compile-verified, matrix operand order checked by reading, not run on hardware
//
#include <hip/hip_runtime.h>

// ---------------------------------------------------------------------------
// PointNet++ part-seg forward for gfx1250 (MI455X).
// All einsums -> v_wmma_f32_16x16x32_f16 GEMMs (f16 in, f32 acc).
// ---------------------------------------------------------------------------

typedef _Float16 half8  __attribute__((ext_vector_type(8)));
typedef _Float16 half16 __attribute__((ext_vector_type(16)));
typedef float    float8 __attribute__((ext_vector_type(8)));

#define WG 256

static __device__ __forceinline__ float sqdist3(float ax, float ay, float az,
                                                float bx, float by, float bz) {
  float dx = ax - bx, dy = ay - by, dz = az - bz;
  return dx * dx + dy * dy + dz * dz;
}

// x [B,3,N] -> xyz [B,N,3]
__global__ void k_transpose_xyz(const float* __restrict__ x, float* __restrict__ xyz,
                                int N, int total) {
  int idx = blockIdx.x * blockDim.x + threadIdx.x;
  if (idx >= total) return;
  int i = idx % N, b = idx / N;
  const float* xb = x + (size_t)b * 3 * N;
  float* o = xyz + (size_t)idx * 3;
  o[0] = xb[i];
  o[1] = xb[N + i];
  o[2] = xb[2 * N + i];
}

// Farthest point sampling: one block per batch, LDS distance array + argmax tree.
// Writes the sampled coordinates (new_xyz) directly.
__global__ void k_fps(const float* __restrict__ xyz, int n, int S,
                      float* __restrict__ new_xyz) {
  __shared__ float dist[2048];
  __shared__ float rv[WG];
  __shared__ int ri[WG];
  int b = blockIdx.x, t = threadIdx.x;
  const float* p = xyz + (size_t)b * n * 3;
  for (int i = t; i < n; i += WG) dist[i] = 1e10f;
  __syncthreads();
  int far = 0;
  for (int s = 0; s < S; ++s) {
    float fx = p[far * 3 + 0], fy = p[far * 3 + 1], fz = p[far * 3 + 2];
    if (t == 0) {
      float* o = new_xyz + ((size_t)b * S + s) * 3;
      o[0] = fx; o[1] = fy; o[2] = fz;
    }
    float best = -1.0f;
    int bi = 0;
    for (int i = t; i < n; i += WG) {
      float d = sqdist3(p[i * 3], p[i * 3 + 1], p[i * 3 + 2], fx, fy, fz);
      float nd = fminf(dist[i], d);
      dist[i] = nd;
      if (nd > best) { best = nd; bi = i; }
    }
    rv[t] = best; ri[t] = bi;
    __syncthreads();
    for (int off = WG / 2; off > 0; off >>= 1) {
      if (t < off && rv[t + off] > rv[t]) { rv[t] = rv[t + off]; ri[t] = ri[t + off]; }
      __syncthreads();
    }
    far = ri[0];
    __syncthreads();
  }
}

// Ball query: first K=32 in-radius indices in ascending order, pad with first.
__global__ void k_ballquery(const float* __restrict__ xyz, const float* __restrict__ new_xyz,
                            int n, int S, float r2, int* __restrict__ gidx, int total) {
  int idx = blockIdx.x * blockDim.x + threadIdx.x;
  if (idx >= total) return;
  int b = idx / S;
  const float* c = new_xyz + (size_t)idx * 3;
  float cx = c[0], cy = c[1], cz = c[2];
  const float* p = xyz + (size_t)b * n * 3;
  int* g = gidx + (size_t)idx * 32;
  int cnt = 0, first = 0;
  for (int i = 0; i < n && cnt < 32; ++i) {
    float d = sqdist3(p[i * 3], p[i * 3 + 1], p[i * 3 + 2], cx, cy, cz);
    if (d <= r2) {
      if (cnt == 0) first = i;
      g[cnt++] = i;
    }
  }
  for (; cnt < 32; ++cnt) g[cnt] = first;
}

// Build grouped GEMM input row: [xyz - center | feats | zero pad to Kp] as f16.
__global__ void k_group_gather(const float* __restrict__ xyz, const float* __restrict__ feats,
                               int fc, const float* __restrict__ new_xyz,
                               const int* __restrict__ gidx, int n, int S, int Kp,
                               _Float16* __restrict__ A, int total) {
  int row = blockIdx.x * blockDim.x + threadIdx.x;
  if (row >= total) return;
  int g = row >> 5;  // K = 32
  int b = g / S;
  int j = gidx[row];
  const float* c = new_xyz + (size_t)g * 3;
  const float* q = xyz + ((size_t)b * n + j) * 3;
  _Float16* a = A + (size_t)row * Kp;
  a[0] = (_Float16)(q[0] - c[0]);
  a[1] = (_Float16)(q[1] - c[1]);
  a[2] = (_Float16)(q[2] - c[2]);
  int w = 3;
  if (fc > 0) {
    const float* f = feats + ((size_t)b * n + j) * fc;
    for (int cc = 0; cc < fc; ++cc) a[w++] = (_Float16)f[cc];
  }
  for (; w < Kp; ++w) a[w] = (_Float16)0.f;
}

// Convert f32 weight [cout,cin] to f16 [Np(=align32 cout) x Kp(=align32 cin)], zero padded.
__global__ void k_convw(const float* __restrict__ W, _Float16* __restrict__ Wh,
                        int cout, int cin, int Kp, int total) {
  int idx = blockIdx.x * blockDim.x + threadIdx.x;
  if (idx >= total) return;
  int c = idx % Kp, o = idx / Kp;
  float v = (o < cout && c < cin) ? W[(size_t)o * cin + c] : 0.f;
  Wh[idx] = (_Float16)v;
}

union U16 { half16 v; half8 h[2]; };

// One wave per 16x16 output tile; K loop in 32-chunks via v_wmma_f32_16x16x32_f16.
// A: [M x Kp] f16 row-major, W: [Np x Kp] f16 row-major (= B^T), Out: [M x Np] f16.
__global__ void k_gemm_wmma(const _Float16* __restrict__ A, const _Float16* __restrict__ W,
                            const float* __restrict__ bias, _Float16* __restrict__ Out,
                            int M, int Kp, int cN, int Np, int relu, int totalWaves) {
  int gw = (blockIdx.x * blockDim.x + threadIdx.x) >> 5;
  if (gw >= totalWaves) return;  // wave-uniform exit, EXEC stays all-ones for WMMA
  int lane = threadIdx.x & 31;
  int ntiles = Np >> 4;
  int mt = gw / ntiles, nt = gw - mt * ntiles;
  int hlf = lane >> 4, l16 = lane & 15;
  const _Float16* arow = A + (size_t)(mt * 16 + l16) * Kp + hlf * 8;
  const _Float16* wrow = W + (size_t)(nt * 16 + l16) * Kp + hlf * 8;
  float8 acc = {};
  for (int kc = 0; kc < Kp; kc += 32) {
    U16 ua, ub;
    // A fragment (16-bit 16x32 layout): two contiguous 8-half (16B) chunks per lane
    ua.h[0] = *(const half8*)(arow + kc);
    ua.h[1] = *(const half8*)(arow + kc + 16);
    ub.h[0] = *(const half8*)(wrow + kc);
    ub.h[1] = *(const half8*)(wrow + kc + 16);
    __builtin_prefetch(arow + kc + 32, 0, 1);
    __builtin_prefetch(wrow + kc + 32, 0, 1);
    acc = __builtin_amdgcn_wmma_f32_16x16x32_f16(
        /*neg_a=*/false, ua.v, /*neg_b=*/false, ub.v,
        /*c_mod=*/(short)0, acc, /*reuse_a=*/false, /*reuse_b=*/false);
  }
  // C/D layout: lanes 0-15 -> M=vgpr, lanes 16-31 -> M=vgpr+8; N = lane&15
  int nn = nt * 16 + l16;
  float bv = (nn < cN) ? bias[nn] : 0.f;
  int mbase = mt * 16 + hlf * 8;
  _Float16* orow = Out + (size_t)mbase * Np + nn;
  for (int v = 0; v < 8; ++v) {
    float xv = acc[v] + bv;
    if (relu) xv = fmaxf(xv, 0.f);
    orow[(size_t)v * Np] = (_Float16)xv;
  }
}

// Max over K=32 group members: H f16 [G*32 x Np] -> feats f32 [G x C]
__global__ void k_maxpool(const _Float16* __restrict__ H, float* __restrict__ feats,
                          int C, int Np, int total) {
  int idx = blockIdx.x * blockDim.x + threadIdx.x;
  if (idx >= total) return;
  int c = idx % C, g = idx / C;
  const _Float16* h = H + ((size_t)g * 32) * Np + c;
  float m = -3.4e38f;
  for (int k = 0; k < 32; ++k) m = fmaxf(m, (float)h[(size_t)k * Np]);
  feats[idx] = m;
}

// FP stage: 3-NN inverse-distance interpolation + concat feats1, emit f16 GEMM row.
__global__ void k_fp_build(const float* __restrict__ xyz1, const float* __restrict__ feats1,
                           int c1, const float* __restrict__ xyz2,
                           const float* __restrict__ feats2, int c2, int n1, int n2,
                           _Float16* __restrict__ A, int Kp, int total) {
  int idx = blockIdx.x * blockDim.x + threadIdx.x;
  if (idx >= total) return;
  int b = idx / n1;
  const float* p = xyz1 + (size_t)idx * 3;
  float px = p[0], py = p[1], pz = p[2];
  const float* q = xyz2 + (size_t)b * n2 * 3;
  float d0 = 3.4e38f, d1 = 3.4e38f, d2 = 3.4e38f;
  int i0 = 0, i1 = 0, i2 = 0;
  for (int j = 0; j < n2; ++j) {
    float d = sqdist3(q[j * 3], q[j * 3 + 1], q[j * 3 + 2], px, py, pz);
    if (d < d0) { d2 = d1; i2 = i1; d1 = d0; i1 = i0; d0 = d; i0 = j; }
    else if (d < d1) { d2 = d1; i2 = i1; d1 = d; i1 = j; }
    else if (d < d2) { d2 = d; i2 = j; }
  }
  float w0 = 1.f / (d0 + 1e-8f), w1 = 1.f / (d1 + 1e-8f), w2 = 1.f / (d2 + 1e-8f);
  float wsum = w0 + w1 + w2;
  w0 /= wsum; w1 /= wsum; w2 /= wsum;
  const float* f0 = feats2 + ((size_t)b * n2 + i0) * c2;
  const float* fa = feats2 + ((size_t)b * n2 + i1) * c2;
  const float* fb = feats2 + ((size_t)b * n2 + i2) * c2;
  _Float16* a = A + (size_t)idx * Kp;
  int w = 0;
  for (int cc = 0; cc < c2; ++cc)
    a[w++] = (_Float16)(w0 * f0[cc] + w1 * fa[cc] + w2 * fb[cc]);
  if (c1 > 0) {
    const float* f1 = feats1 + (size_t)idx * c1;
    for (int cc = 0; cc < c1; ++cc) a[w++] = (_Float16)f1[cc];
  }
  for (; w < Kp; ++w) a[w] = (_Float16)0.f;
}

// f16 GEMM output [R x Np] -> f32 feats [R x C]
__global__ void k_h2f(const _Float16* __restrict__ H, float* __restrict__ feats,
                      int C, int Np, int total) {
  int idx = blockIdx.x * blockDim.x + threadIdx.x;
  if (idx >= total) return;
  int c = idx % C, r = idx / C;
  feats[idx] = (float)H[(size_t)r * Np + c];
}

// Seg head input: [128-ch feats | 16 one-hot | pad to 160] f16 per point.
__global__ void k_seg_build(const float* __restrict__ feats, const int* __restrict__ cls,
                            _Float16* __restrict__ A, int total) {
  int idx = blockIdx.x * blockDim.x + threadIdx.x;
  if (idx >= total) return;
  int b = idx >> 11;  // N = 2048
  const float* f = feats + (size_t)idx * 128;
  _Float16* a = A + (size_t)idx * 160;
  for (int c = 0; c < 128; ++c) a[c] = (_Float16)f[c];
  int cl = cls[b];
  for (int c = 0; c < 16; ++c) a[128 + c] = (_Float16)((c == cl) ? 1.f : 0.f);
  for (int c = 144; c < 160; ++c) a[c] = (_Float16)0.f;
}

// H f16 [(B*N) x 64] -> out f32 [B,50,N]
__global__ void k_seg_emit(const _Float16* __restrict__ H, float* __restrict__ out, int total) {
  int idx = blockIdx.x * blockDim.x + threadIdx.x;
  if (idx >= total) return;
  int n = idx & 2047;
  int rest = idx >> 11;
  int o = rest % 50;
  int b = rest / 50;
  out[idx] = (float)H[((size_t)b * 2048 + n) * 64 + o];
}

static inline int align32i(int x) { return (x + 31) & ~31; }

extern "C" void kernel_launch(void* const* d_in, const int* in_sizes, int n_in,
                              void* d_out, int out_size, void* d_ws, size_t ws_size,
                              hipStream_t stream) {
  (void)in_sizes; (void)n_in; (void)out_size; (void)ws_size;
  const int B = 32, N = 2048, K = 32;
  const int nL[5] = {2048, 1024, 256, 64, 16};
  const int cL[5] = {0, 64, 128, 256, 512};
  const float rad[4] = {0.1f, 0.2f, 0.4f, 0.8f};
  const int sa_ch[4][3] = {{32, 48, 64}, {64, 96, 128}, {128, 196, 256}, {256, 384, 512}};
  const int fp_ch[4][2] = {{256, 256}, {256, 128}, {128, 128}, {128, 128}};

  char* base = (char*)d_ws;
  size_t off = 0;
  auto alloc = [&](size_t bytes) -> void* {
    off = (off + 255) & ~(size_t)255;
    void* p = base + off;
    off += bytes;
    return p;
  };

  const float* x = (const float*)d_in[0];
  const int* cls = (const int*)d_in[1];

  const float* saW[4][3]; const float* saB[4][3];
  const float* fpW[4][2]; const float* fpB[4][2];
  int wi = 2;
  for (int l = 0; l < 4; ++l)
    for (int j = 0; j < 3; ++j) {
      saW[l][j] = (const float*)d_in[wi++];
      saB[l][j] = (const float*)d_in[wi++];
    }
  for (int l = 0; l < 4; ++l)
    for (int j = 0; j < 2; ++j) {
      fpW[l][j] = (const float*)d_in[wi++];
      fpB[l][j] = (const float*)d_in[wi++];
    }
  const float* segW = (const float*)d_in[wi++];
  const float* segB = (const float*)d_in[wi++];

  // ---- convert all weights to padded f16 once ----
  _Float16* saWh[4][3]; _Float16* fpWh[4][2]; _Float16* segWh;
  auto convw = [&](const float* W, int cout, int cin) -> _Float16* {
    int Kp = align32i(cin), Np = align32i(cout);
    _Float16* Wh = (_Float16*)alloc((size_t)Np * Kp * sizeof(_Float16));
    int total = Np * Kp;
    k_convw<<<(total + 255) / 256, 256, 0, stream>>>(W, Wh, cout, cin, Kp, total);
    return Wh;
  };
  for (int l = 0; l < 4; ++l) {
    int cin = 3 + cL[l];
    for (int j = 0; j < 3; ++j) {
      saWh[l][j] = convw(saW[l][j], sa_ch[l][j], cin);
      cin = sa_ch[l][j];
    }
  }
  for (int l = 0; l < 4; ++l) {
    int c2 = (l == 0) ? cL[4] : fp_ch[l - 1][1];
    int cin = c2 + cL[3 - l];
    for (int j = 0; j < 2; ++j) {
      fpWh[l][j] = convw(fpW[l][j], fp_ch[l][j], cin);
      cin = fp_ch[l][j];
    }
  }
  segWh = convw(segW, 50, 144);

  // ---- persistent point/feature buffers ----
  float* xyzL[5]; float* featL[5];
  xyzL[0] = (float*)alloc((size_t)B * N * 3 * sizeof(float));
  featL[0] = nullptr;
  for (int l = 1; l < 5; ++l) {
    xyzL[l] = (float*)alloc((size_t)B * nL[l] * 3 * sizeof(float));
    featL[l] = (float*)alloc((size_t)B * nL[l] * cL[l] * sizeof(float));
  }
  int* gidx = (int*)alloc((size_t)B * 1024 * 32 * sizeof(int));
  // ping-pong GEMM buffers sized for SA1: 1,048,576 rows x 64-half stride = 128 MiB each
  size_t bufBytes = (size_t)B * 1024 * 32 * 64 * sizeof(_Float16);
  _Float16* bufA = (_Float16*)alloc(bufBytes);
  _Float16* bufB = (_Float16*)alloc(bufBytes);

  auto gemm = [&](const _Float16* Ain, const _Float16* Wh, const float* bias,
                  _Float16* Out, int M, int Kp, int cN, int relu) -> int {
    int Np = align32i(cN);
    int totalWaves = (M / 16) * (Np / 16);
    int blocks = (totalWaves * 32 + 255) / 256;
    k_gemm_wmma<<<blocks, 256, 0, stream>>>(Ain, Wh, bias, Out, M, Kp, cN, Np, relu,
                                            totalWaves);
    return Np;
  };

  // ---- input transpose ----
  {
    int total = B * N;
    k_transpose_xyz<<<(total + 255) / 256, 256, 0, stream>>>(x, xyzL[0], N, total);
  }

  // ---- SA levels ----
  for (int l = 0; l < 4; ++l) {
    int n = nL[l], S = nL[l + 1], fc = cL[l];
    k_fps<<<B, WG, 0, stream>>>(xyzL[l], n, S, xyzL[l + 1]);
    int totBS = B * S;
    k_ballquery<<<(totBS + 255) / 256, 256, 0, stream>>>(
        xyzL[l], xyzL[l + 1], n, S, rad[l] * rad[l], gidx, totBS);
    int cin = 3 + fc, Kp = align32i(cin);
    int M = B * S * K;
    k_group_gather<<<(M + 255) / 256, 256, 0, stream>>>(
        xyzL[l], featL[l], fc, xyzL[l + 1], gidx, n, S, Kp, bufA, M);
    _Float16* cur = bufA; _Float16* nxt = bufB; int curKp = Kp;
    for (int j = 0; j < 3; ++j) {
      int Np = gemm(cur, saWh[l][j], saB[l][j], nxt, M, curKp, sa_ch[l][j], 1);
      _Float16* t = cur; cur = nxt; nxt = t;
      curKp = Np;
    }
    int C = cL[l + 1];
    int tot = B * S * C;
    k_maxpool<<<(tot + 255) / 256, 256, 0, stream>>>(cur, featL[l + 1], C, curKp, tot);
  }

  // ---- FP levels ----
  float* fpfeat[4];
  for (int i = 0; i < 4; ++i) {
    int n2 = nL[4 - i], n1 = nL[3 - i];
    const float* xyz2 = xyzL[4 - i];
    const float* xyz1 = xyzL[3 - i];
    const float* f2 = (i == 0) ? featL[4] : fpfeat[i - 1];
    int c2 = (i == 0) ? cL[4] : fp_ch[i - 1][1];
    const float* f1 = featL[3 - i];
    int c1 = cL[3 - i];
    int cin = c2 + c1, Kp = align32i(cin);
    int M = B * n1;
    k_fp_build<<<(M + 255) / 256, 256, 0, stream>>>(xyz1, f1, c1, xyz2, f2, c2, n1, n2,
                                                    bufA, Kp, M);
    _Float16* cur = bufA; _Float16* nxt = bufB; int curKp = Kp;
    for (int j = 0; j < 2; ++j) {
      int Np = gemm(cur, fpWh[i][j], fpB[i][j], nxt, M, curKp, fp_ch[i][j], 1);
      _Float16* t = cur; cur = nxt; nxt = t;
      curKp = Np;
    }
    int co = fp_ch[i][1];
    fpfeat[i] = (float*)alloc((size_t)M * co * sizeof(float));
    int tot = M * co;
    k_h2f<<<(tot + 255) / 256, 256, 0, stream>>>(cur, fpfeat[i], co, curKp, tot);
  }

  // ---- segmentation head ----
  {
    int M = B * N;
    k_seg_build<<<(M + 255) / 256, 256, 0, stream>>>(fpfeat[3], cls, bufA, M);
    gemm(bufA, segWh, segB, bufB, M, 160, 50, 0);
    int tot = B * 50 * N;
    k_seg_emit<<<(tot + 255) / 256, 256, 0, stream>>>(bufB, (float*)d_out, tot);
  }
}
